// ContextGraphModel_73864847557026
// MI455X (gfx1250) — matrix operand order repaired and checked by reading, not compiled, not run on hardware
//
#include <hip/hip_runtime.h>
#include <hip/hip_bf16.h>
#include <math.h>

typedef __bf16 bf16_t;
typedef __attribute__((ext_vector_type(8)))  __bf16 bf8v;
typedef __attribute__((ext_vector_type(16))) __bf16 v16bf;
typedef __attribute__((ext_vector_type(8)))  float  v8f;

#define GEMM_BM 128
#define GEMM_BN 128
#define GEMM_KT 32
#define LDSR    40   // padded LDS row stride in halves (breaks bank conflicts)

// ---------------------------------------------------------------------------
// Tiled bf16 WMMA GEMM: C[M,N] = act(A[M,K] @ B[K,N] + bias), fp32 accumulate.
// A is row-major bf16 (lda=K). Bt is B transposed: N x K row-major bf16.
// 256 threads = 8 waves (4 along M x 2 along N); wave tile 32x64 = 2x4 frags.
// ---------------------------------------------------------------------------
template<int ACT>  // 0=none, 1=sigmoid, 2=tanh
__global__ __launch_bounds__(256)
void gemm_bf16_wmma(const bf16_t* __restrict__ A,
                    const bf16_t* __restrict__ Bt,
                    const float*  __restrict__ bias,
                    float*        __restrict__ C,
                    int Mdim, int N, int K)
{
    __shared__ bf16_t As[GEMM_BM * LDSR];
    __shared__ bf16_t Bs[GEMM_BN * LDSR];

    const int tid  = threadIdx.x;
    const int lane = tid & 31;
    const int wid  = tid >> 5;
    const int wm   = wid & 3;     // wave row  (4 waves * 32 rows = 128)
    const int wn   = wid >> 2;    // wave col  (2 waves * 64 cols = 128)
    const int bm   = blockIdx.x * GEMM_BM;
    const int bn   = blockIdx.y * GEMM_BN;

    const int lr = lane & 15;     // 0..15
    const int lh = lane >> 4;     // 0/1

    v8f acc[2][4];
#pragma unroll
    for (int mi = 0; mi < 2; ++mi)
#pragma unroll
        for (int ni = 0; ni < 4; ++ni)
#pragma unroll
            for (int i = 0; i < 8; ++i) acc[mi][ni][i] = 0.0f;

    for (int k0 = 0; k0 < K; k0 += GEMM_KT) {
        __syncthreads();
        // Cooperative tile load: 128x32 halves each for A and Bt, 8-half b128 chunks.
#pragma unroll
        for (int s = 0; s < 2; ++s) {
            int seg = tid + s * 256;        // 0..511
            int r   = seg >> 2;             // tile row 0..127
            int c   = (seg & 3) * 8;        // half offset 0,8,16,24
            bf8v av = {};
            int gr = bm + r;
            if (gr < Mdim) av = *(const bf8v*)(A + (size_t)gr * K + k0 + c);
            *(bf8v*)(As + r * LDSR + c) = av;
            int gn = bn + r;                // N is a multiple of 128 -> no guard
            bf8v bv = *(const bf8v*)(Bt + (size_t)gn * K + k0 + c);
            *(bf8v*)(Bs + r * LDSR + c) = bv;
        }
        __syncthreads();

        // A fragment (16x32): halves 0..7 = K kb..kb+7, 8..15 = K kb+16..kb+23
        v16bf af[2], bfv[4];
        const int kb = lh * 8;
#pragma unroll
        for (int mi = 0; mi < 2; ++mi) {
            int r = wm * 32 + mi * 16 + lr;
            bf8v lo = *(const bf8v*)(As + r * LDSR + kb);
            bf8v hi = *(const bf8v*)(As + r * LDSR + kb + 16);
#pragma unroll
            for (int i = 0; i < 8; ++i) { af[mi][i] = lo[i]; af[mi][i + 8] = hi[i]; }
        }
        // B fragment (32x16): halves 0..7 = K ks..ks+7, 8..15 = K ks+16..ks+23
        const int ks = lh * 8;
#pragma unroll
        for (int ni = 0; ni < 4; ++ni) {
            int r = wn * 64 + ni * 16 + lr;   // row of Bt = column of B
            bf8v lo = *(const bf8v*)(Bs + r * LDSR + ks);
            bf8v hi = *(const bf8v*)(Bs + r * LDSR + ks + 16);
#pragma unroll
            for (int i = 0; i < 8; ++i) { bfv[ni][i] = lo[i]; bfv[ni][i + 8] = hi[i]; }
        }

#pragma unroll
        for (int mi = 0; mi < 2; ++mi)
#pragma unroll
            for (int ni = 0; ni < 4; ++ni)
                acc[mi][ni] = __builtin_amdgcn_wmma_f32_16x16x32_bf16(
                    false, af[mi], false, bfv[ni], (short)0, acc[mi][ni], false, false);
    }

    // Epilogue: D vgpr r -> M = r + 8*lh, N = lr  (ISA 16x16 f32 C/D layout)
#pragma unroll
    for (int mi = 0; mi < 2; ++mi) {
#pragma unroll
        for (int ni = 0; ni < 4; ++ni) {
            int col = bn + wn * 64 + ni * 16 + lr;
            float bv = bias ? bias[col] : 0.0f;
#pragma unroll
            for (int r = 0; r < 8; ++r) {
                int row = bm + wm * 32 + mi * 16 + lh * 8 + r;
                if (row < Mdim) {
                    float v = acc[mi][ni][r] + bv;
                    if (ACT == 1)      v = 1.0f / (1.0f + expf(-v));
                    else if (ACT == 2) v = tanhf(v);
                    C[(size_t)row * N + col] = v;
                }
            }
        }
    }
}

// ---------------------------------------------------------------------------
// Support kernels
// ---------------------------------------------------------------------------
__global__ void k_transpose_bf16(const float* __restrict__ W, bf16_t* __restrict__ Wt,
                                 int K, int N)
{
    size_t idx = (size_t)blockIdx.x * blockDim.x + threadIdx.x;
    if (idx >= (size_t)K * N) return;
    int n = (int)(idx / K), k = (int)(idx % K);
    Wt[idx] = (bf16_t)W[(size_t)k * N + n];
}

__global__ void k_fill_zero(float* __restrict__ p, size_t n)
{
    size_t i = (size_t)blockIdx.x * blockDim.x + threadIdx.x;
    if (i < n) p[i] = 0.0f;
}

// Build x0 = [label_emb, masked-max type_emb] in bf16 (V x 256)
__global__ void k_init_embed(const int* __restrict__ label_ids,
                             const int* __restrict__ type_ids,
                             const float* __restrict__ mask,
                             const float* __restrict__ label_tab,
                             const float* __restrict__ type_tab,
                             bf16_t* __restrict__ x0, int V)
{
    int v = blockIdx.x;
    int j = threadIdx.x;
    if (v >= V) return;
    float out;
    if (j < 128) {
        out = label_tab[(size_t)label_ids[v] * 128 + j];
    } else {
        int jj = j - 128;
        float mx = -3.4e38f;
        for (int t = 0; t < 10; ++t) {
            float e = type_tab[(size_t)type_ids[v * 10 + t] * 128 + jj]
                      + (1.0f - mask[v * 10 + t]) * -10000000.0f;
            mx = fmaxf(mx, e);
        }
        out = mx;
    }
    x0[(size_t)v * 256 + j] = (bf16_t)out;
}

// dst[r*stride + colOff + c] = bf16(src[r*256 + c])
__global__ void k_f32_to_bf16_block(const float* __restrict__ src, bf16_t* __restrict__ dst,
                                    int dstStride, int colOff, int V)
{
    size_t idx = (size_t)blockIdx.x * blockDim.x + threadIdx.x;
    if (idx >= (size_t)V * 256) return;
    int r = (int)(idx >> 8), c = (int)(idx & 255);
    dst[(size_t)r * dstStride + colOff + c] = (bf16_t)src[idx];
}

__global__ void k_copy_bf16_block(const bf16_t* __restrict__ src, bf16_t* __restrict__ dst,
                                  int dstStride, int colOff, int V)
{
    size_t idx = (size_t)blockIdx.x * blockDim.x + threadIdx.x;
    if (idx >= (size_t)V * 256) return;
    int r = (int)(idx >> 8), c = (int)(idx & 255);
    dst[(size_t)r * dstStride + colOff + c] = src[idx];
}

// dst slot = bf16(r_gate * h), r_gate = gates[:, 0:256]
__global__ void k_rh_block(const float* __restrict__ gates, const float* __restrict__ h,
                           bf16_t* __restrict__ dst, int dstStride, int colOff, int V)
{
    size_t idx = (size_t)blockIdx.x * blockDim.x + threadIdx.x;
    if (idx >= (size_t)V * 256) return;
    int r = (int)(idx >> 8), c = (int)(idx & 255);
    dst[(size_t)r * dstStride + colOff + c] = (bf16_t)(gates[(size_t)r * 512 + c] * h[idx]);
}

// h = u*h + (1-u)*c ; hb = bf16(h). u = gates[:, 256:512]
__global__ void k_gru_update(const float* __restrict__ gates, const float* __restrict__ cbuf,
                             float* __restrict__ h, bf16_t* __restrict__ hb, int V)
{
    size_t idx = (size_t)blockIdx.x * blockDim.x + threadIdx.x;
    if (idx >= (size_t)V * 256) return;
    int r = (int)(idx >> 8), c = (int)(idx & 255);
    float u  = gates[(size_t)r * 512 + 256 + c];
    float nv = u * h[idx] + (1.0f - u) * cbuf[idx];
    h[idx]  = nv;
    hb[idx] = (bf16_t)nv;
}

// msg[dst_node] += t[src_node] ; one block (256 threads) per edge, coalesced atomics
__global__ void k_scatter_add(const float* __restrict__ t, const int* __restrict__ adj_e,
                              int gcol, float* __restrict__ msg, int M)
{
    int m = blockIdx.x;
    int j = threadIdx.x;
    if (m >= M) return;
    int a = adj_e[(size_t)m * 2 + gcol];        // gather node
    int b = adj_e[(size_t)m * 2 + (1 - gcol)];  // target node
    atomicAdd(msg + (size_t)b * 256 + j, t[(size_t)a * 256 + j]);
}

// ---------------------------------------------------------------------------
extern "C" void kernel_launch(void* const* d_in, const int* in_sizes, int n_in,
                              void* d_out, int out_size, void* d_ws, size_t ws_size,
                              hipStream_t stream)
{
    const int V = 50000, H = 256, M = 100000;
    const int timesteps[4] = {3, 1, 3, 1};
    const int dsz[4]       = {H, 2 * H, H, 3 * H};
    const size_t VH = (size_t)V * 256;

    const int*   label_ids = (const int*)d_in[0];
    const int*   type_ids  = (const int*)d_in[1];
    const float* type_mask = (const float*)d_in[2];
    const int*   adj       = (const int*)d_in[3];
    const float* label_tab = (const float*)d_in[4];
    const float* type_tab  = (const float*)d_in[5];
    const float* w_init    = (const float*)d_in[6];
    const float* edge_w    = (const float*)d_in[7];
    const float *wg[4], *bg[4], *wc[4], *bc[4];
    for (int l = 0; l < 4; ++l) {
        wg[l] = (const float*)d_in[8  + 4 * l];
        bg[l] = (const float*)d_in[9  + 4 * l];
        wc[l] = (const float*)d_in[10 + 4 * l];
        bc[l] = (const float*)d_in[11 + 4 * l];
    }

    // ---- workspace bump allocator (256B aligned) ----
    char* p = (char*)d_ws;
    auto alloc = [&](size_t bytes) -> void* {
        void* q = (void*)p;
        p += (bytes + 255) & ~(size_t)255;
        return q;
    };
    bf16_t* winit_t = (bf16_t*)alloc((size_t)256 * 256 * 2);
    bf16_t* edge_t  = (bf16_t*)alloc((size_t)32 * 256 * 256 * 2);
    bf16_t *wg_t[4], *wc_t[4];
    for (int l = 0; l < 4; ++l) {
        wg_t[l] = (bf16_t*)alloc((size_t)512 * (dsz[l] + H) * 2);
        wc_t[l] = (bf16_t*)alloc((size_t)256 * (dsz[l] + H) * 2);
    }
    float*  hS    = (float*)alloc(VH * 4);            // current state, fp32
    float*  msg   = (float*)alloc(VH * 4);            // message accumulator (alias: cbuf)
    float*  gates = (float*)alloc((size_t)V * 512 * 4); // gates (alias: tbuf)
    bf16_t* hb    = (bf16_t*)alloc(VH * 2);           // bf16 copy of state
    bf16_t* s0b   = (bf16_t*)alloc(VH * 2);           // states_per_layer[0]
    bf16_t* s1b   = (bf16_t*)alloc(VH * 2);           // states_per_layer[1]
    bf16_t* xg    = (bf16_t*)alloc((size_t)V * 1024 * 2); // concat input buffer
    float* tbuf = gates;  // edge transform buffer lives in GRU-phase-only gates buf
    float* cbuf = msg;    // candidate buffer lives in msg (dead by candidate GEMM)

    const unsigned VHB = (unsigned)((VH + 255) / 256);

    // ---- weight prep: fp32 -> bf16, transposed to N x K (column-major B) ----
    auto tpose = [&](const float* W, bf16_t* Wt, int K, int N) {
        size_t tot = (size_t)K * N;
        k_transpose_bf16<<<(unsigned)((tot + 255) / 256), 256, 0, stream>>>(W, Wt, K, N);
    };
    tpose(w_init, winit_t, 256, 256);
    for (int i = 0; i < 32; ++i)
        tpose(edge_w + (size_t)i * 65536, edge_t + (size_t)i * 65536, 256, 256);
    for (int l = 0; l < 4; ++l) {
        tpose(wg[l], wg_t[l], dsz[l] + H, 512);
        tpose(wc[l], wc_t[l], dsz[l] + H, 256);
    }

    auto gemm = [&](const bf16_t* A, const bf16_t* Bt, const float* bias, float* C,
                    int Md, int N, int K, int act) {
        dim3 g((Md + GEMM_BM - 1) / GEMM_BM, N / GEMM_BN);
        if (act == 0)      gemm_bf16_wmma<0><<<g, 256, 0, stream>>>(A, Bt, bias, C, Md, N, K);
        else if (act == 1) gemm_bf16_wmma<1><<<g, 256, 0, stream>>>(A, Bt, bias, C, Md, N, K);
        else               gemm_bf16_wmma<2><<<g, 256, 0, stream>>>(A, Bt, bias, C, Md, N, K);
    };

    // ---- init: h = [label_e, max type_e] @ w_init ----
    k_init_embed<<<V, 256, 0, stream>>>(label_ids, type_ids, type_mask,
                                        label_tab, type_tab, xg, V);
    gemm(xg, winit_t, nullptr, hS, V, 256, 256, 0);
    k_f32_to_bf16_block<<<VHB, 256, 0, stream>>>(hS, hb, 256, 0, V);
    k_copy_bf16_block<<<VHB, 256, 0, stream>>>(hb, s0b, 256, 0, V);

    // ---- propagation layers ----
    for (int l = 0; l < 4; ++l) {
        const int d = dsz[l];
        const int stride = d + H;
        for (int s = 0; s < timesteps[l]; ++s) {
            k_fill_zero<<<VHB, 256, 0, stream>>>(msg, VH);
            // message passing: transform densely over V (cheaper than over M), then scatter
            for (int dir = 0; dir < 8; ++dir) {
                const bf16_t* Wt = edge_t + (size_t)(l * 8 + dir) * 65536;
                gemm(hb, Wt, nullptr, tbuf, V, 256, 256, 0);
                int e    = dir & 3;
                int gcol = (dir < 4) ? 0 : 1;  // fwd: gather src->tgt ; bwd: gather tgt->src
                k_scatter_add<<<M, 256, 0, stream>>>(tbuf, adj + (size_t)e * M * 2,
                                                     gcol, msg, M);
            }
            // build concat [res..., msg, h] in bf16
            int off = 0;
            if (l == 1) {
                k_copy_bf16_block<<<VHB, 256, 0, stream>>>(s0b, xg, stride, 0, V);
                off = 256;
            } else if (l == 3) {
                k_copy_bf16_block<<<VHB, 256, 0, stream>>>(s0b, xg, stride, 0, V);
                k_copy_bf16_block<<<VHB, 256, 0, stream>>>(s1b, xg, stride, 256, V);
                off = 512;
            }
            k_f32_to_bf16_block<<<VHB, 256, 0, stream>>>(msg, xg, stride, off, V);
            k_copy_bf16_block<<<VHB, 256, 0, stream>>>(hb, xg, stride, d, V);
            // gates = sigmoid([x,h] @ wg + bg)
            gemm(xg, wg_t[l], bg[l], gates, V, 512, stride, 1);
            // overwrite h-slot with r*h, candidate = tanh([x, r*h] @ wc + bc)
            k_rh_block<<<VHB, 256, 0, stream>>>(gates, hS, xg, stride, d, V);
            gemm(xg, wc_t[l], bc[l], cbuf, V, 256, stride, 2);
            // h = u*h + (1-u)*c
            k_gru_update<<<VHB, 256, 0, stream>>>(gates, cbuf, hS, hb, V);
        }
        if (l == 0)
            k_copy_bf16_block<<<VHB, 256, 0, stream>>>(hb, s1b, 256, 0, V);
    }

    hipMemcpyAsync(d_out, hS, VH * 4, hipMemcpyDeviceToDevice, stream);
}
